// RNN_wavefunction_6708738916401
// MI455X (gfx1250) — matrix operand order repaired and checked by reading, not compiled, not run on hardware
//
#include <hip/hip_runtime.h>

typedef __attribute__((ext_vector_type(16))) _Float16 v16h;
typedef __attribute__((ext_vector_type(8)))  _Float16 v8h;
typedef __attribute__((ext_vector_type(8)))  float    v8f;

#define NSTEP 64
#define U 64
#define ROWS_PER_BLOCK 64
#define THREADS 128

__device__ __forceinline__ v16h ld_frag2(const _Float16* p0, const _Float16* p1) {
    v8h lo = *(const v8h*)p0;
    v8h hi = *(const v8h*)p1;
    return __builtin_shufflevector(lo, hi, 0,1,2,3,4,5,6,7,8,9,10,11,12,13,14,15);
}

__device__ __forceinline__ float fast_tanh(float x) {
#if __has_builtin(__builtin_amdgcn_tanhf)
    return __builtin_amdgcn_tanhf(x);          // v_tanh_f32 (CDNA5 trans op)
#elif __has_builtin(__builtin_amdgcn_tanh_f32)
    return __builtin_amdgcn_tanh_f32(x);
#else
    // tanh(x) = 1 - 2/(exp2(2*log2e*x)+1); saturates correctly at +-inf
    float e = __builtin_amdgcn_exp2f(x * 2.8853900817779268f);
    return 1.0f - 2.0f * __builtin_amdgcn_rcpf(e + 1.0f);
#endif
}

#define WMMA_F16(Afrag, Bfrag, Cacc) \
    __builtin_amdgcn_wmma_f32_16x16x32_f16(false, (Afrag), false, (Bfrag), (short)0, (Cacc), false, false)

__global__ __launch_bounds__(THREADS)
void rnn_wavefn_kernel(const float* __restrict__ sample,
                       const float* __restrict__ W_ih0, const float* __restrict__ b_ih0,
                       const float* __restrict__ W_hh0, const float* __restrict__ b_hh0,
                       const float* __restrict__ W_ih1, const float* __restrict__ b_ih1,
                       const float* __restrict__ W_hh1, const float* __restrict__ b_hh1,
                       const float* __restrict__ Wd,    const float* __restrict__ bd,
                       float* __restrict__ out)
{
    // Row-major f16 weights: [0]=W_hh0 [1]=W_ih1 [2]=W_hh1 (each 64x64)
    __shared__ __attribute__((aligned(32))) _Float16 Wlds[3 * 4096];
    __shared__ __attribute__((aligned(32))) _Float16 Wdpad[16 * 64];    // rows 0,1 = Wd, rest 0
    __shared__ __attribute__((aligned(32))) _Float16 Hst[2 * 4 * 1024]; // [stage][wave][16x64]
    __shared__ unsigned int MaskB[ROWS_PER_BLOCK * 2];

    const int tid  = threadIdx.x;
    const int wv   = tid >> 5;
    const int ln   = tid & 31;
    const int lo16 = ln & 15;
    const int hi   = ln >> 4;     // lane half
    const int hi8  = hi * 8;

    // ---------------- setup: stage weights as f16, build sample bitmasks ----------------
    for (int i = tid; i < 4096; i += THREADS) {
        Wlds[i]          = (_Float16)W_hh0[i];
        Wlds[4096 + i]   = (_Float16)W_ih1[i];
        Wlds[8192 + i]   = (_Float16)W_hh1[i];
    }
    for (int i = tid; i < 16 * 64; i += THREADS) {
        int r = i >> 6, c = i & 63;
        Wdpad[i] = (_Float16)((r < 2) ? Wd[r * 64 + c] : 0.0f);
    }
    for (int i = tid; i < 2 * 4 * 1024; i += THREADS) Hst[i] = (_Float16)0.0f;
    {
        int row  = tid >> 1;          // 64 rows per block
        int half = tid & 1;           // 32 steps per half
        const float* sp = sample
            + (size_t)(blockIdx.x * ROWS_PER_BLOCK + row) * (NSTEP * 2)
            + half * 64 + 1;          // one-hot component [*,*,1]
        unsigned int bits = 0;
        #pragma unroll
        for (int k = 0; k < 32; ++k)
            bits |= (sp[k * 2] > 0.5f) ? (1u << k) : 0u;
        MaskB[row * 2 + half] = bits;
    }
    __syncthreads();

    // ---------------- per-lane constants ----------------
    float w0[4], w1[4], bias0[4], bias1[4];
    #pragma unroll
    for (int t = 0; t < 4; ++t) {
        int j = t * 16 + lo16;             // feature column this lane owns in tile t
        w0[t]    = W_ih0[j * 2 + 0];
        w1[t]    = W_ih0[j * 2 + 1];
        bias0[t] = b_ih0[j] + b_hh0[j];
        bias1[t] = b_ih1[j] + b_hh1[j];
    }
    float bdv = (lo16 < 2) ? bd[lo16] : 0.0f;

    unsigned long long cur[8];
    unsigned int prevbit[8];
    #pragma unroll
    for (int r = 0; r < 8; ++r) {
        int rb = wv * 16 + hi8 + r;
        unsigned long long mlo = MaskB[rb * 2 + 0];
        unsigned long long mhi = MaskB[rb * 2 + 1];
        cur[r]     = mlo | (mhi << 32);
        prevbit[r] = 0;
    }

    // ---------------- register-resident B fragments (weights) ----------------
    // B (32x16 f16): lane holds column N=lo16, 16 contiguous K values (K = 32c + 16*hi ..)
    const int boff = 16 * hi;
    v16h bWhh0[2][4], bWih1[2][4], bWhh1[2][4], bWd[2];
    #pragma unroll
    for (int c = 0; c < 2; ++c) {
        #pragma unroll
        for (int t = 0; t < 4; ++t) {
            int off = (t * 16 + lo16) * 64 + 32 * c + boff;
            bWhh0[c][t] = *(const v16h*)&Wlds[off];
            bWih1[c][t] = *(const v16h*)&Wlds[4096 + off];
            bWhh1[c][t] = *(const v16h*)&Wlds[8192 + off];
        }
        bWd[c] = *(const v16h*)&Wdpad[lo16 * 64 + 32 * c + boff];
    }

    _Float16* h0s = &Hst[wv * 1024];
    _Float16* h1s = &Hst[4096 + wv * 1024];
    const int aM = lo16;      // A-fragment row (M) this lane owns
    const int aK = 16 * hi;   // A-fragment K base within chunk

    float prod[8];
    #pragma unroll
    for (int r = 0; r < 8; ++r) prod[r] = 1.0f;

    // ---------------- recurrence over 64 steps ----------------
    for (int s = 0; s < NSTEP; ++s) {
        // ---- layer 0: pre0 = (x@W_ih0^T + b_ih0 + b_hh0) + h0@W_hh0^T ----
        v8f acc0[4];
        #pragma unroll
        for (int t = 0; t < 4; ++t) {
            #pragma unroll
            for (int r = 0; r < 8; ++r) {
                float xi = (s > 0) ? (prevbit[r] ? w1[t] : w0[t]) : 0.0f;
                acc0[t][r] = bias0[t] + xi;
            }
        }
        #pragma unroll
        for (int c = 0; c < 2; ++c) {
            v16h a = ld_frag2(&h0s[aM * 64 + 32 * c + aK],
                              &h0s[aM * 64 + 32 * c + aK + 8]);
            #pragma unroll
            for (int t = 0; t < 4; ++t)
                acc0[t] = WMMA_F16(a, bWhh0[c][t], acc0[t]);
        }
        // h0n = tanh(pre0) -> stage (row-major f16) for A-fragment reads
        #pragma unroll
        for (int t = 0; t < 4; ++t)
            #pragma unroll
            for (int r = 0; r < 8; ++r)
                h0s[(hi8 + r) * 64 + t * 16 + lo16] = (_Float16)fast_tanh(acc0[t][r]);

        // ---- layer 1: pre1 = h0n@W_ih1^T + h1@W_hh1^T + biases ----
        v8f acc1[4];
        #pragma unroll
        for (int t = 0; t < 4; ++t)
            #pragma unroll
            for (int r = 0; r < 8; ++r)
                acc1[t][r] = bias1[t];
        #pragma unroll
        for (int c = 0; c < 2; ++c) {
            v16h a0 = ld_frag2(&h0s[aM * 64 + 32 * c + aK],
                               &h0s[aM * 64 + 32 * c + aK + 8]);
            v16h a1 = ld_frag2(&h1s[aM * 64 + 32 * c + aK],
                               &h1s[aM * 64 + 32 * c + aK + 8]);
            #pragma unroll
            for (int t = 0; t < 4; ++t) {
                acc1[t] = WMMA_F16(a0, bWih1[c][t], acc1[t]);
                acc1[t] = WMMA_F16(a1, bWhh1[c][t], acc1[t]);
            }
        }
        #pragma unroll
        for (int t = 0; t < 4; ++t)
            #pragma unroll
            for (int r = 0; r < 8; ++r)
                h1s[(hi8 + r) * 64 + t * 16 + lo16] = (_Float16)fast_tanh(acc1[t][r]);

        // ---- head: logits = h1n @ Wd^T + bd (as one padded WMMA tile) ----
        v8f accL;
        #pragma unroll
        for (int r = 0; r < 8; ++r) accL[r] = bdv;
        #pragma unroll
        for (int c = 0; c < 2; ++c) {
            v16h a1 = ld_frag2(&h1s[aM * 64 + 32 * c + aK],
                               &h1s[aM * 64 + 32 * c + aK + 8]);
            accL = WMMA_F16(a1, bWd[c], accL);
        }
        // sigmoid form of the 2-way softmax:
        //   p_own = 1 / (1 + exp(l_other - l_own))
        // lanes lo16==0/1 hold the two logits per row; pick prob of observed spin
        #pragma unroll
        for (int r = 0; r < 8; ++r) {
            float l   = accL[r];
            float o   = __shfl_xor(l, 1, 32);
            float e   = __builtin_amdgcn_exp2f((o - l) * 1.4426950408889634f);
            float p   = __builtin_amdgcn_rcpf(1.0f + e);
            unsigned int bit = (unsigned int)cur[r] & 1u;
            bool take = ((int)bit == lo16);     // only lanes lo16<2 meaningful
            prod[r] *= take ? p : 1.0f;
            prevbit[r] = bit;
            cur[r] >>= 1;
        }
    }

    // combine the two per-outcome partial products, lane lo16==0 writes out
    float tot[8];
    #pragma unroll
    for (int r = 0; r < 8; ++r) tot[r] = prod[r] * __shfl_xor(prod[r], 1, 32);
    if (lo16 == 0) {
        int base = blockIdx.x * ROWS_PER_BLOCK + wv * 16 + hi8;
        #pragma unroll
        for (int r = 0; r < 8; ++r) out[base + r] = tot[r];
    }
}

extern "C" void kernel_launch(void* const* d_in, const int* in_sizes, int n_in,
                              void* d_out, int out_size, void* d_ws, size_t ws_size,
                              hipStream_t stream) {
    (void)in_sizes; (void)n_in; (void)d_ws; (void)ws_size; (void)out_size;
    const float* sample = (const float*)d_in[0];
    const float* W_ih0  = (const float*)d_in[1];
    const float* b_ih0  = (const float*)d_in[2];
    const float* W_hh0  = (const float*)d_in[3];
    const float* b_hh0  = (const float*)d_in[4];
    const float* W_ih1  = (const float*)d_in[5];
    const float* b_ih1  = (const float*)d_in[6];
    const float* W_hh1  = (const float*)d_in[7];
    const float* b_hh1  = (const float*)d_in[8];
    const float* Wd     = (const float*)d_in[9];
    const float* bd     = (const float*)d_in[10];
    float* out = (float*)d_out;

    dim3 grid(16384 / ROWS_PER_BLOCK);   // 256 blocks
    dim3 block(THREADS);                 // 4 waves
    rnn_wavefn_kernel<<<grid, block, 0, stream>>>(
        sample, W_ih0, b_ih0, W_hh0, b_hh0,
        W_ih1, b_ih1, W_hh1, b_hh1, Wd, bd, out);
}
